// LPKTNet_26285199852324
// MI455X (gfx1250) — matrix operand order repaired and checked by reading, not compiled, not run on hardware
//
#include <hip/hip_runtime.h>
#include <hip/hip_bf16.h>

typedef _Float16 h8   __attribute__((ext_vector_type(8)));
typedef _Float16 v16h __attribute__((ext_vector_type(16)));
typedef float    v8f  __attribute__((ext_vector_type(8)));

#define HS  132   // h32 row stride (floats): 4-bank shift/row, 16B aligned
#define WST 136   // f16 weight row stride (halves): 4-bank shift/row, 16B aligned

__device__ __forceinline__ v16h cat16(h8 x, h8 y) {
    return __builtin_shufflevector(x, y, 0,1,2,3,4,5,6,7,8,9,10,11,12,13,14,15);
}
__device__ __forceinline__ v8f v8f_zero() {
    v8f z = {0.f,0.f,0.f,0.f,0.f,0.f,0.f,0.f}; return z;
}
__device__ __forceinline__ float sigmoidf(float x) { return 1.f / (1.f + expf(-x)); }

// gfx1250 async copy: global -> LDS, 16B per lane, tracked by ASYNCcnt.
__device__ __forceinline__ void async_g2l_b128(unsigned lds_off, const void* gptr) {
    asm volatile("global_load_async_to_lds_b128 %0, %1, off"
                 :: "v"(lds_off), "v"((unsigned long long)(uintptr_t)gptr)
                 : "memory");
}
__device__ __forceinline__ void wait_async0() {
    asm volatile("s_wait_asynccnt 0" ::: "memory");
}

// ---------------------------------------------------------------------------
// K1: all_learning L[b*128+s][j] = b1[j] + [e_emb, at_emb, a*1_50] @ W1[j].T
// 8 (b,s) rows per block so W1 rows are read once per 8 outputs.
// ---------------------------------------------------------------------------
__global__ __launch_bounds__(128) void k1_learning(
    const int* __restrict__ e_data, const int* __restrict__ at_data,
    const float* __restrict__ a_data, const float* __restrict__ e_w,
    const float* __restrict__ at_w, const float* __restrict__ W1,
    const float* __restrict__ b1, float* __restrict__ L)
{
    __shared__ float e8[8][128];
    __shared__ float at8[8][128];
    __shared__ float a8[8];
    const int tid = threadIdx.x;
    const int m0  = blockIdx.x * 8;
    for (int r = 0; r < 8; ++r) {
        int m = m0 + r;
        e8[r][tid]  = e_w [(size_t)e_data [m] * 128 + tid];
        at8[r][tid] = at_w[(size_t)at_data[m] * 128 + tid];
    }
    if (tid < 8) a8[tid] = a_data[m0 + tid];
    __syncthreads();

    const int j = tid;
    const float* w = W1 + (size_t)j * 306;
    float acc[8];
    float bj = b1[j];
    #pragma unroll
    for (int r = 0; r < 8; ++r) acc[r] = bj;
    for (int k = 0; k < 128; ++k) {
        float wv = w[k];
        #pragma unroll
        for (int r = 0; r < 8; ++r) acc[r] += e8[r][k] * wv;
    }
    for (int k = 0; k < 128; ++k) {
        float wv = w[128 + k];
        #pragma unroll
        for (int r = 0; r < 8; ++r) acc[r] += at8[r][k] * wv;
    }
    float asum = 0.f;
    for (int k = 0; k < 50; ++k) asum += w[256 + k];
    #pragma unroll
    for (int r = 0; r < 8; ++r)
        L[(size_t)(m0 + r) * 128 + j] = acc[r] + a8[r] * asum;
}

// ---------------------------------------------------------------------------
// K2: pack WbigT[n][k] (512x512, f16):
//   n<128 : p2  = W2[n][k]                       for k<384
//   n<256 : p3  = W3[n-128][k]                   for k<384
//   n<384 : gfb = W4c = W4[n-256][256+(k-128)]   for 128<=k<256
//   n<512 : yp  = W5a = W5[n-384][k-384]         for k>=384
// ---------------------------------------------------------------------------
__global__ __launch_bounds__(128) void k2_pack_w(
    const float* __restrict__ W2, const float* __restrict__ W3,
    const float* __restrict__ W4, const float* __restrict__ W5,
    _Float16* __restrict__ Wt)
{
    const int n = blockIdx.x;
    for (int k = threadIdx.x; k < 512; k += 128) {
        float v = 0.f;
        if (n < 128)      { if (k < 384)              v = W2[(size_t)n * 512 + k]; }
        else if (n < 256) { if (k < 384)              v = W3[(size_t)(n-128) * 512 + k]; }
        else if (n < 384) { if (k >= 128 && k < 256)  v = W4[(size_t)(n-256) * 384 + 256 + (k-128)]; }
        else              { if (k >= 384)             v = W5[(size_t)(n-384) * 256 + (k-384)]; }
        Wt[(size_t)n * 512 + k] = (_Float16)v;
    }
}

// ---------------------------------------------------------------------------
// K3: pack X[m][0:512] = [L[m-1] (0 at t=0) | it_emb | L[m] | e_emb]  (f16)
// ---------------------------------------------------------------------------
__global__ __launch_bounds__(128) void k3_pack_x(
    const int* __restrict__ e_data, const int* __restrict__ it_data,
    const float* __restrict__ e_w, const float* __restrict__ it_w,
    const float* __restrict__ L, _Float16* __restrict__ X)
{
    const int m = blockIdx.x;
    const int t = m & 127;
    const int j = threadIdx.x;
    _Float16* xm = X + (size_t)m * 512;
    xm[j]         = (t > 0) ? (_Float16)L[(size_t)(m-1) * 128 + j] : (_Float16)0.f;
    xm[128 + j]   = (_Float16)it_w[(size_t)it_data[m] * 128 + j];
    xm[256 + j]   = (_Float16)L[(size_t)m * 128 + j];
    xm[384 + j]   = (_Float16)e_w[(size_t)e_data[m] * 128 + j];
}

// ---------------------------------------------------------------------------
// K4: OUT[8192, 512] = X[8192,512] @ WbigT[512,512]^T   (f16 WMMA, f32 acc)
// block: 256 thr (8 waves), tile M=128 N=128, K staged 64 at a time in LDS
// via async global->LDS copies (ASYNCcnt). B tiles preloaded 4 at a time so
// WMMAs issue back-to-back behind a single dscnt wait.
// ---------------------------------------------------------------------------
__global__ __launch_bounds__(256) void k4_gemm(
    const _Float16* __restrict__ X, const _Float16* __restrict__ Wt,
    float* __restrict__ OUT)
{
    __shared__ _Float16 As[128 * 72];
    __shared__ _Float16 Bs[128 * 72];
    const int tid  = threadIdx.x;
    const int row0 = blockIdx.x * 128;
    const int n0   = blockIdx.y * 128;
    const int wave = tid >> 5;
    const int lane = tid & 31;
    const int lo   = lane & 15;
    const int hi   = lane >> 4;
    const unsigned asBase = (unsigned)(uintptr_t)(void*)As;   // LDS byte offset
    const unsigned bsBase = (unsigned)(uintptr_t)(void*)Bs;

    v8f acc[8];
    #pragma unroll
    for (int nt = 0; nt < 8; ++nt) acc[nt] = v8f_zero();

    for (int kc = 0; kc < 8; ++kc) {
        const int k0 = kc * 64;
        __syncthreads();   // previous tile fully consumed before overwrite
        for (int i = tid; i < 1024; i += 256) {   // 128 rows x 8 chunks of 8 halves
            int r = i >> 3, cg = i & 7;
            async_g2l_b128(asBase + (unsigned)(r * 72 + cg * 8) * 2,
                           X  + (size_t)(row0 + r) * 512 + k0 + cg * 8);
        }
        for (int i = tid; i < 1024; i += 256) {
            int r = i >> 3, cg = i & 7;
            async_g2l_b128(bsBase + (unsigned)(r * 72 + cg * 8) * 2,
                           Wt + (size_t)(n0 + r)   * 512 + k0 + cg * 8);
        }
        wait_async0();
        __syncthreads();
        #pragma unroll
        for (int kt = 0; kt < 2; ++kt) {
            const int kb = kt * 32;
            // A operand: lane<16 -> row lo, K {kb..kb+7, kb+16..kb+23}; lane>=16 -> +8
            const _Float16* pa = As + (wave * 16 + lo) * 72 + kb + hi * 8;
            v16h a = cat16(*(const h8*)pa, *(const h8*)(pa + 16));
            #pragma unroll
            for (int g = 0; g < 2; ++g) {
                v16h bm[4];
                #pragma unroll
                for (int q = 0; q < 4; ++q) {
                    const _Float16* pb = Bs + ((g * 4 + q) * 16 + lo) * 72 + kb + hi * 8;
                    bm[q] = cat16(*(const h8*)pb, *(const h8*)(pb + 16));
                }
                #pragma unroll
                for (int q = 0; q < 4; ++q)
                    acc[g * 4 + q] = __builtin_amdgcn_wmma_f32_16x16x32_f16(
                        false, a, false, bm[q], (short)0, acc[g * 4 + q], false, false);
            }
        }
    }
    #pragma unroll
    for (int nt = 0; nt < 8; ++nt) {
        #pragma unroll
        for (int v = 0; v < 8; ++v) {
            int r = row0 + wave * 16 + v + 8 * hi;   // C layout: VGPR v -> M=v(+8), lane -> N
            int c = n0 + nt * 16 + lo;
            OUT[(size_t)r * 512 + c] = acc[nt][v];
        }
    }
}

// ---------------------------------------------------------------------------
// K5: sequential LPKT scan. 1 workgroup per batch element, 512 thr (16 waves).
// LDS: h (272x132 f32), 5 x 128x136 f16 weights, small vectors. ~316 KB/WGP.
// Per step: VALU matvecs for LG/gf_bias, 544 v_wmma_f32_16x16x32_f16 for
// h_pre @ W4a.T (B tiles preloaded in groups of 4), gated in-place h update,
// weighted row-reduction for h_tilde, and the y readout.
// ---------------------------------------------------------------------------
#define K5_SMEM (272*HS*4 + 5*128*WST*2 + (272+272+128+128+128+512+128)*4)

__global__ __launch_bounds__(512) void k5_recur(
    const int*  __restrict__ e_data, const float* __restrict__ qmat,
    const float* __restrict__ h0,
    const float* __restrict__ W2, const float* __restrict__ W3,
    const float* __restrict__ W4, const float* __restrict__ W5,
    const float* __restrict__ b2, const float* __restrict__ b3,
    const float* __restrict__ b4, const float* __restrict__ b5,
    const float* __restrict__ OUT, float* __restrict__ pred)
{
    extern __shared__ char smem[];
    float*     h32  = (float*)smem;                       // 272*132 f32
    _Float16*  w4a  = (_Float16*)(h32 + 272 * HS);        // each 128*136 f16
    _Float16*  w2d  = w4a + 128 * WST;
    _Float16*  w3d  = w2d + 128 * WST;
    _Float16*  w4b  = w3d + 128 * WST;
    _Float16*  w5b  = w4b + 128 * WST;
    float*     qe_s = (float*)(w5b + 128 * WST);          // 272
    float*     qn_s = qe_s + 272;                         // 272
    float*     htl  = qn_s + 272;                         // 128
    float*     LG_s = htl + 128;                          // 128
    float*     gfb_s= LG_s + 128;                         // 128
    float*     part = gfb_s + 128;                        // 512
    float*     red  = part + 512;                         // 128

    const int b   = blockIdx.x;
    const int tid = threadIdx.x;
    const int wave = tid >> 5, lane = tid & 31, lo = lane & 15, hi = lane >> 4;

    // ---- init: stage step-loop weights (f32 -> f16) and h0 into LDS ----
    for (int i = tid; i < 16384; i += 512) {
        int r = i >> 7, c = i & 127;
        w4a[r * WST + c] = (_Float16)W4[(size_t)r * 384 + c];          // W4[:, :128]
        w2d[r * WST + c] = (_Float16)W2[(size_t)r * 512 + 384 + c];    // W2[:, 384:]
        w3d[r * WST + c] = (_Float16)W3[(size_t)r * 512 + 384 + c];    // W3[:, 384:]
        w4b[r * WST + c] = (_Float16)W4[(size_t)r * 384 + 128 + c];    // W4[:,128:256]
        w5b[r * WST + c] = (_Float16)W5[(size_t)r * 256 + 128 + c];    // W5[:,128:]
    }
    for (int i = tid; i < 272 * 128; i += 512) {
        int r = i >> 7, c = i & 127;
        h32[r * HS + c] = (r < 257) ? h0[(size_t)r * 128 + c] : 0.f;
    }
    {   // h_tilde_0 = q_all[:,0] @ h0
        int e0 = e_data[b * 128];
        for (int r = tid; r < 272; r += 512)
            qn_s[r] = (r < 257) ? qmat[(size_t)e0 * 257 + r] : 0.f;
    }
    if (tid == 0) pred[b * 128] = 0.f;
    __syncthreads();
    {
        int col = tid & 127, p = tid >> 7; float s = 0.f;
        for (int r = p; r < 272; r += 4) s += qn_s[r] * h32[r * HS + col];
        part[tid] = s;
    }
    __syncthreads();
    if (tid < 128) htl[tid] = part[tid] + part[tid+128] + part[tid+256] + part[tid+384];
    __syncthreads();

    // ---- the scan: 127 sequential steps ----
    for (int i = 0; i < 127; ++i) {
        const int m  = b * 128 + i;
        const int ei = e_data[m], en = e_data[m + 1];
        for (int r = tid; r < 272; r += 512) {
            qe_s[r] = (r < 257) ? qmat[(size_t)ei * 257 + r] : 0.f;
            qn_s[r] = (r < 257) ? qmat[(size_t)en * 257 + r] : 0.f;
        }
        if (tid < 128) {                       // lg, gamma_l, LG
            int j = tid;
            float p2 = OUT[(size_t)m * 512 + j]       + b2[j];
            float p3 = OUT[(size_t)m * 512 + 128 + j] + b3[j];
            for (int k = 0; k < 128; ++k) {
                float hv = htl[k];
                p2 += hv * (float)w2d[j * WST + k];
                p3 += hv * (float)w3d[j * WST + k];
            }
            LG_s[j] = sigmoidf(p3) * (tanhf(p2) + 1.f) * 0.5f;
        }
        __syncthreads();
        if (tid < 128) {                       // gf_bias = LG@W4b.T + it@W4c.T + b4
            int j = tid;
            float g = OUT[(size_t)m * 512 + 256 + j] + b4[j];
            for (int k = 0; k < 128; ++k) g += LG_s[k] * (float)w4b[j * WST + k];
            gfb_s[j] = g;
        }
        __syncthreads();

        // GEMM h_pre @ W4a.T + gated in-place update. Each wave owns its own
        // 16 rows of h (wave0 also takes tail tile 16) -> no cross-wave hazard.
        for (int mt = wave; mt < 17; mt += 16) {
            v8f acc[8];
            #pragma unroll
            for (int nt = 0; nt < 8; ++nt) acc[nt] = v8f_zero();
            #pragma unroll
            for (int kt = 0; kt < 4; ++kt) {
                const float* pa = h32 + (mt * 16 + lo) * HS + kt * 32 + hi * 8;
                v16h a;
                #pragma unroll
                for (int e = 0; e < 8; ++e) a[e]     = (_Float16)pa[e];
                #pragma unroll
                for (int e = 0; e < 8; ++e) a[8 + e] = (_Float16)pa[16 + e];
                #pragma unroll
                for (int g = 0; g < 2; ++g) {
                    v16h bm[4];
                    #pragma unroll
                    for (int q = 0; q < 4; ++q) {
                        const _Float16* pb = w4a + ((g * 4 + q) * 16 + lo) * WST + kt * 32 + hi * 8;
                        bm[q] = cat16(*(const h8*)pb, *(const h8*)(pb + 16));
                    }
                    #pragma unroll
                    for (int q = 0; q < 4; ++q)
                        acc[g * 4 + q] = __builtin_amdgcn_wmma_f32_16x16x32_f16(
                            false, a, false, bm[q], (short)0, acc[g * 4 + q], false, false);
                }
            }
            #pragma unroll
            for (int nt = 0; nt < 8; ++nt) {
                int col = nt * 16 + lo;
                float lgv = LG_s[col], gb = gfb_s[col];
                #pragma unroll
                for (int v = 0; v < 8; ++v) {
                    int row = mt * 16 + v + 8 * hi;
                    float g  = sigmoidf(acc[nt][v] + gb);
                    float hp = h32[row * HS + col];
                    h32[row * HS + col] = qe_s[row] * lgv + g * hp;
                }
            }
        }
        __syncthreads();

        {   // h_tilde = q_next @ h
            int col = tid & 127, p = tid >> 7; float s = 0.f;
            for (int r = p; r < 272; r += 4) s += qn_s[r] * h32[r * HS + col];
            part[tid] = s;
        }
        __syncthreads();
        if (tid < 128) htl[tid] = part[tid] + part[tid+128] + part[tid+256] + part[tid+384];
        __syncthreads();
        if (tid < 128) {                       // y readout
            int j = tid;
            float y = OUT[(size_t)(m + 1) * 512 + 384 + j] + b5[j];
            for (int k = 0; k < 128; ++k) y += htl[k] * (float)w5b[j * WST + k];
            red[j] = sigmoidf(y);
        }
        __syncthreads();
        if (tid == 0) {
            float s = 0.f;
            for (int k = 0; k < 128; ++k) s += red[k];
            pred[m + 1] = s * (1.f / 128.f);
        }
        __syncthreads();
    }
}

// ---------------------------------------------------------------------------
extern "C" void kernel_launch(void* const* d_in, const int* in_sizes, int n_in,
                              void* d_out, int out_size, void* d_ws, size_t ws_size,
                              hipStream_t stream) {
    const int*   e_data  = (const int*)  d_in[0];
    const int*   at_data = (const int*)  d_in[1];
    const float* a_data  = (const float*)d_in[2];
    const int*   it_data = (const int*)  d_in[3];
    const float* qmat    = (const float*)d_in[4];
    const float* at_w    = (const float*)d_in[5];
    const float* it_w    = (const float*)d_in[6];
    const float* e_w     = (const float*)d_in[7];
    const float* h0      = (const float*)d_in[8];
    const float* W1      = (const float*)d_in[9];
    const float* b1      = (const float*)d_in[10];
    const float* W2      = (const float*)d_in[11];
    const float* b2      = (const float*)d_in[12];
    const float* W3      = (const float*)d_in[13];
    const float* b3      = (const float*)d_in[14];
    const float* W4      = (const float*)d_in[15];
    const float* b4      = (const float*)d_in[16];
    const float* W5      = (const float*)d_in[17];
    const float* b5      = (const float*)d_in[18];
    float* pred = (float*)d_out;

    // workspace layout (all 16B aligned)
    char* ws = (char*)d_ws;
    float*    L   = (float*)ws;                                  //  4 MB: 64*128*128 f32
    _Float16* X   = (_Float16*)(ws + 4194304);                   //  8 MB: 8192*512 f16
    _Float16* Wt  = (_Float16*)(ws + 4194304 + 8388608);         // .5 MB: 512*512 f16
    float*    OUT = (float*)(ws + 4194304 + 8388608 + 524288);   // 16 MB: 8192*512 f32

    k1_learning<<<1024, 128, 0, stream>>>(e_data, at_data, a_data, e_w, at_w, W1, b1, L);
    k2_pack_w  <<<512,  128, 0, stream>>>(W2, W3, W4, W5, Wt);
    k3_pack_x  <<<8192, 128, 0, stream>>>(e_data, it_data, e_w, it_w, L, X);
    k4_gemm    <<<dim3(64, 4), 256, 0, stream>>>(X, Wt, OUT);

    (void)hipFuncSetAttribute(reinterpret_cast<const void*>(k5_recur),
                              hipFuncAttributeMaxDynamicSharedMemorySize, (int)K5_SMEM);
    k5_recur<<<64, 512, K5_SMEM, stream>>>(e_data, qmat, h0, W2, W3, W4, W5,
                                           b2, b3, b4, b5, OUT, pred);
    (void)in_sizes; (void)n_in; (void)out_size; (void)ws_size;
}
